// Model_13606456394557
// MI455X (gfx1250) — matrix operand order repaired and checked by reading, not compiled, not run on hardware
//
#include <hip/hip_runtime.h>

// ---------- CDNA5 WMMA types ----------
typedef __attribute__((ext_vector_type(16))) __bf16 v16bf;
typedef __attribute__((ext_vector_type(8)))  __bf16 v8bf;
typedef __attribute__((ext_vector_type(8)))  float  v8f;

#define WAVES_PER_BLOCK 4

__device__ __forceinline__ v8f wmma_bf16(v16bf a, v16bf b, v8f c) {
  // D = A(16x32 bf16) * B(32x16 bf16) + C(16x16 f32)
  return __builtin_amdgcn_wmma_f32_16x16x32_bf16(
      /*neg_a=*/false, a, /*neg_b=*/false, b,
      /*c_mod=*/(short)0, c, /*reuse_a=*/false, /*reuse_b=*/false);
}

// gfx1250 hardware tanh (TRANS op). Builtin if available, else raw ISA with a
// v_nop to respect the 1-op TRANS result hazard.
__device__ __forceinline__ float tanh_hw(float x) {
#if __has_builtin(__builtin_amdgcn_tanhf)
  return __builtin_amdgcn_tanhf(x);
#else
  float y;
  asm("v_tanh_f32 %0, %1\n\tv_nop" : "=v"(y) : "v"(x));
  return y;
#endif
}
__device__ __forceinline__ float sigmoid_hw(float x) {
  return 0.5f * tanh_hw(0.5f * x) + 0.5f;   // 1/(1+e^-x) = (1+tanh(x/2))/2
}

// ---------- fragment builders ----------
// B fragment (32x16, bf16): B[k][n] = W[n][k], W row-major [nOut][ldw] fp32.
// Lane layout: n = lane&15; lanes 0-15 hold K = kb+{0..7, 16..23}, lanes 16-31 hold K = kb+{8..15, 24..31}.
__device__ __forceinline__ v16bf bfrag_global(const float* __restrict__ W,
                                              int ldw, int nb, int kb) {
  const int lane = threadIdx.x & 31;
  const float* p = W + (nb + (lane & 15)) * ldw + kb + ((lane >> 4) << 3);
  const float4* q = (const float4*)p;
  float4 w0 = q[0], w1 = q[1], w2 = q[4], w3 = q[5];
  v16bf f;
  f[0]=(__bf16)w0.x; f[1]=(__bf16)w0.y; f[2]=(__bf16)w0.z; f[3]=(__bf16)w0.w;
  f[4]=(__bf16)w1.x; f[5]=(__bf16)w1.y; f[6]=(__bf16)w1.z; f[7]=(__bf16)w1.w;
  f[8]=(__bf16)w2.x; f[9]=(__bf16)w2.y; f[10]=(__bf16)w2.z; f[11]=(__bf16)w2.w;
  f[12]=(__bf16)w3.x; f[13]=(__bf16)w3.y; f[14]=(__bf16)w3.z; f[15]=(__bf16)w3.w;
  return f;
}

// A fragment (16x32, bf16) straight from global x (row-major [N][64] fp32).
// Lane layout: row M = lane&15; lanes 0-15 hold K = kb+{0..7, 16..23},
// lanes 16-31 hold K = kb+{8..15, 24..31}.
__device__ __forceinline__ v16bf afrag_x(const float* __restrict__ x,
                                         int rowBase, int kb, int N) {
  const int lane = threadIdx.x & 31;
  int row = rowBase + (lane & 15);
  row = row < N ? row : N - 1;                 // tail-safe (store is masked)
  const int half = lane >> 4;
  const float* p = x + row * 64 + kb + half * 8;
  const float4* q0 = (const float4*)p;
  const float4* q1 = (const float4*)(p + 16);
  float4 c0 = q0[0], c1 = q0[1], c2 = q1[0], c3 = q1[1];
  v16bf f;
  f[0]=(__bf16)c0.x; f[1]=(__bf16)c0.y; f[2]=(__bf16)c0.z; f[3]=(__bf16)c0.w;
  f[4]=(__bf16)c1.x; f[5]=(__bf16)c1.y; f[6]=(__bf16)c1.z; f[7]=(__bf16)c1.w;
  f[8]=(__bf16)c2.x; f[9]=(__bf16)c2.y; f[10]=(__bf16)c2.z; f[11]=(__bf16)c2.w;
  f[12]=(__bf16)c3.x; f[13]=(__bf16)c3.y; f[14]=(__bf16)c3.z; f[15]=(__bf16)c3.w;
  return f;
}

// A fragment from a per-wave LDS h-tile (row-major bf16 [16][128]).
__device__ __forceinline__ v16bf afrag_lds(const __bf16* h, int kb) {
  const int lane = threadIdx.x & 31;
  const int r = lane & 15;
  const int half = lane >> 4;
  v8bf lo = *(const v8bf*)(h + r * 128 + kb + half * 8);        // ds_load_b128
  v8bf hi = *(const v8bf*)(h + r * 128 + kb + 16 + half * 8);   // ds_load_b128
  v16bf f;
#pragma unroll
  for (int e = 0; e < 8; ++e) { f[e] = lo[e]; f[e + 8] = hi[e]; }
  return f;
}

// tanh(D-tile) -> LDS h-tile (row-major bf16). D layout: VGPR r holds
// (M = r + (lane>>4)*8, N = nb + (lane&15)).
__device__ __forceinline__ void store_tanh_tile(__bf16* h, int nb, v8f acc) {
  const int lane = threadIdx.x & 31;
  const int c = lane & 15;
  const int mh = (lane >> 4) << 3;
#pragma unroll
  for (int r = 0; r < 8; ++r)
    h[(mh + r) * 128 + nb + c] = (__bf16)tanh_hw(acc[r]);
}

__device__ __forceinline__ void lds_fence() {
  asm volatile("s_wait_dscnt 0" ::: "memory");
}

// ---------- fused 2-layer RNN step + FC ----------
__global__ __launch_bounds__(WAVES_PER_BLOCK * 32)
void rnn_fused_kernel(const float* __restrict__ x,
                      const float* __restrict__ w_ih0, const float* __restrict__ w_hh0,
                      const float* __restrict__ b_ih0, const float* __restrict__ b_hh0,
                      const float* __restrict__ w_ih1, const float* __restrict__ w_hh1,
                      const float* __restrict__ b_ih1, const float* __restrict__ b_hh1,
                      const float* __restrict__ w_fc,  const float* __restrict__ b_fc,
                      float* __restrict__ out, int N) {
  extern __shared__ __align__(16) unsigned char smem_raw[];
  const int lane = threadIdx.x & 31;
  const int wave = threadIdx.x >> 5;

  // LDS map: [0,32KB) layer-1 weight fragments (32 frags x 32 lanes x 32B),
  //          then per-wave h1 / h2 bf16 scratch tiles (4KB each).
  unsigned* w1lds = (unsigned*)smem_raw;
  __bf16* h1 = (__bf16*)(smem_raw + 32768) + wave * 2048;
  __bf16* h2 = (__bf16*)(smem_raw + 32768 + WAVES_PER_BLOCK * 4096) + wave * 2048;

  // One-time: stage W_ih1 as bf16 B-fragments in LDS (wave w owns k-block w).
#pragma unroll
  for (int j = 0; j < 8; ++j) {
    v16bf f = bfrag_global(w_ih1, 128, j * 16, wave * 32);
    *(v16bf*)(w1lds + ((wave * 8 + j) * 32 + lane) * 8) = f;
  }
  __syncthreads();

  // Register-resident weights: W_ih0 (2 k-blocks x 8 n-blocks), W_fc (4 k-blocks).
  v16bf B0[2][8];
#pragma unroll
  for (int kb = 0; kb < 2; ++kb)
#pragma unroll
    for (int j = 0; j < 8; ++j)
      B0[kb][j] = bfrag_global(w_ih0, 64, j * 16, kb * 32);
  v16bf Bf[4];
#pragma unroll
  for (int kb = 0; kb < 4; ++kb)
    Bf[kb] = bfrag_global(w_fc, 128, 0, kb * 32);

  // Constant recurrent terms: rec = rowsum(W_hh) + b_hh + b_ih (h_prev = ones).
  float rec0v[8], rec1v[8];
#pragma unroll
  for (int j = 0; j < 8; ++j) {
    const int n = j * 16 + (lane & 15);
    float s0 = b_ih0[n] + b_hh0[n];
    float s1 = b_ih1[n] + b_hh1[n];
    const float4* r0 = (const float4*)(w_hh0 + n * 128);
    const float4* r1 = (const float4*)(w_hh1 + n * 128);
#pragma unroll
    for (int k = 0; k < 32; ++k) {
      float4 a = r0[k]; s0 += (a.x + a.y) + (a.z + a.w);
      float4 b = r1[k]; s1 += (b.x + b.y) + (b.z + b.w);
    }
    rec0v[j] = s0; rec1v[j] = s1;
  }
  const float bfcv = b_fc[lane & 15];

  const int tiles = (N + 15) >> 4;
  const int gw = blockIdx.x * WAVES_PER_BLOCK + wave;
  const int nw = gridDim.x * WAVES_PER_BLOCK;

  for (int t = gw; t < tiles; t += nw) {
    const int rowBase = t << 4;

    // ---- layer 0: h1 = tanh(x @ W_ih0^T + rec0) : 16 WMMA ----
    v16bf a0 = afrag_x(x, rowBase, 0, N);
    v16bf a1 = afrag_x(x, rowBase, 32, N);
#pragma unroll
    for (int j = 0; j < 8; ++j) {
      v8f acc;
#pragma unroll
      for (int r = 0; r < 8; ++r) acc[r] = rec0v[j];
      acc = wmma_bf16(a0, B0[0][j], acc);
      acc = wmma_bf16(a1, B0[1][j], acc);
      store_tanh_tile(h1, j * 16, acc);
    }
    lds_fence();

    // ---- layer 1: h2 = tanh(h1 @ W_ih1^T + rec1) : 32 WMMA ----
    v16bf ah[4];
#pragma unroll
    for (int kb = 0; kb < 4; ++kb) ah[kb] = afrag_lds(h1, kb * 32);
#pragma unroll
    for (int j = 0; j < 8; ++j) {
      v8f acc;
#pragma unroll
      for (int r = 0; r < 8; ++r) acc[r] = rec1v[j];
#pragma unroll
      for (int kb = 0; kb < 4; ++kb) {
        v16bf b = *(const v16bf*)(w1lds + ((kb * 8 + j) * 32 + lane) * 8);
        acc = wmma_bf16(ah[kb], b, acc);
      }
      store_tanh_tile(h2, j * 16, acc);
    }
    lds_fence();

    // ---- fc: out = sigmoid(h2 @ W_fc^T + b_fc) : 4 WMMA ----
    v8f acc;
#pragma unroll
    for (int r = 0; r < 8; ++r) acc[r] = bfcv;
#pragma unroll
    for (int kb = 0; kb < 4; ++kb)
      acc = wmma_bf16(afrag_lds(h2, kb * 32), Bf[kb], acc);

    const int c = lane & 15;
    const int mh = (lane >> 4) << 3;
#pragma unroll
    for (int r = 0; r < 8; ++r) {
      const int row = rowBase + mh + r;
      if (row < N) out[row * 16 + c] = sigmoid_hw(acc[r]);
    }
  }
}

// ---------- launch ----------
extern "C" void kernel_launch(void* const* d_in, const int* in_sizes, int n_in,
                              void* d_out, int out_size, void* d_ws, size_t ws_size,
                              hipStream_t stream) {
  (void)n_in; (void)out_size; (void)d_ws; (void)ws_size;
  const float* x     = (const float*)d_in[0];
  const float* w_ih0 = (const float*)d_in[1];
  const float* w_hh0 = (const float*)d_in[2];
  const float* b_ih0 = (const float*)d_in[3];
  const float* b_hh0 = (const float*)d_in[4];
  const float* w_ih1 = (const float*)d_in[5];
  const float* w_hh1 = (const float*)d_in[6];
  const float* b_ih1 = (const float*)d_in[7];
  const float* b_hh1 = (const float*)d_in[8];
  const float* w_fc  = (const float*)d_in[9];
  const float* b_fc  = (const float*)d_in[10];

  const int N = in_sizes[0] / 64;
  const int tiles = (N + 15) / 16;
  int blocks = (tiles + WAVES_PER_BLOCK - 1) / WAVES_PER_BLOCK;
  if (blocks > 1024) blocks = 1024;
  if (blocks < 1) blocks = 1;

  const size_t smem = 32768 + (size_t)WAVES_PER_BLOCK * 4096 * 2;  // 64 KB
  rnn_fused_kernel<<<blocks, WAVES_PER_BLOCK * 32, smem, stream>>>(
      x, w_ih0, w_hh0, b_ih0, b_hh0, w_ih1, w_hh1, b_ih1, b_hh1,
      w_fc, b_fc, (float*)d_out, N);
}
